// PewForecastModel_498216206990
// MI455X (gfx1250) — compile-verified, matrix operand order checked by reading, not run on hardware
//
#include <hip/hip_runtime.h>
#include <hip/hip_bf16.h>

// ---------------------------------------------------------------------------
// PewForecastModel: 2-layer weather-conditioned LSTM, fused persistent kernel.
// B=512, T=168, H=512, 4H=2048, WD=4.
//
//  * 32 blocks x 16 batch rows; batch rows independent through the recurrence
//    -> no inter-block sync, each block loops t=0..167 locally.
//  * h-state [16 x 1024] bf16 in LDS (cols 0..511 = h1, 512..1023 = h2),
//    c-state in VGPRs. Weights bf16, pre-packed to WMMA B-fragment layout.
//  * B-fragment addressing: one per-lane base pointer, compile-time immediate
//    offsets -> loads clause up, scheduler pipelines them under the WMMAs.
//  * Weight stream uses non-temporal loads (CDNA5 TH=NT): 6 MB/step/block has
//    no near-cache reuse; keep it L2-resident without thrashing the WGP$.
// ---------------------------------------------------------------------------

typedef __bf16    v16bf  __attribute__((ext_vector_type(16)));
typedef float     v8f    __attribute__((ext_vector_type(8)));
typedef unsigned  nuint4 __attribute__((ext_vector_type(4)));   // native 128-bit vec

union Frag {
    v16bf  v;
    nuint4 q[2];
    unsigned short s[16];
};
static_assert(sizeof(v16bf) == 32, "v16bf must be 32B");
static_assert(sizeof(nuint4) == 16, "nuint4 must be 16B");

__device__ __forceinline__ unsigned short f2bf(float f) {
    unsigned u = __builtin_bit_cast(unsigned, f);
    unsigned r = 0x7FFFu + ((u >> 16) & 1u);        // round-to-nearest-even
    return (unsigned short)((u + r) >> 16);
}
__device__ __forceinline__ float bf2f(unsigned short b) {
    return __builtin_bit_cast(float, (unsigned)b << 16);
}
__device__ __forceinline__ unsigned pack_bf16x2(float a, float b) {
    return (unsigned)f2bf(a) | ((unsigned)f2bf(b) << 16);
}

__device__ __forceinline__ float sigmoid_f(float x) {
    return 1.0f / (1.0f + __expf(-x));
}
__device__ __forceinline__ float tanh_f(float x) {
#if __has_builtin(__builtin_amdgcn_tanhf)
    return __builtin_amdgcn_tanhf(x);               // v_tanh_f32 (gfx1250 TRANS)
#else
    float e = __expf(2.0f * x);
    return (e - 1.0f) / (e + 1.0f);
#endif
}

// ---------------------------------------------------------------------------
// Pack a (K0+K1) x 2048 fp32 weight matrix (rows 0..K0-1 from src0, rows
// K0.. from src1) into bf16 WMMA B-fragment layout:
//   dword index = (((kChunk*128 + nTile)*32) + lane)*8 + r
//   element (r, e, lane): K = kChunk*32 + 16*(r>>2) + 2*(r&3) + 8*(lane>>4) + e
//                         N = nTile*16 + (lane&15)
// ---------------------------------------------------------------------------
__global__ void pack_wmma_b(const float* __restrict__ src0, int K0,
                            const float* __restrict__ src1,
                            unsigned* __restrict__ dst, long total_dwords) {
    long idx = (long)blockIdx.x * blockDim.x + threadIdx.x;
    if (idx >= total_dwords) return;
    int r    = (int)(idx & 7);
    int lane = (int)((idx >> 3) & 31);
    int nT   = (int)((idx >> 8) & 127);
    int kc   = (int)(idx >> 15);
    int g    = lane >> 4;
    int ln   = lane & 15;
    int K    = kc * 32 + 16 * (r >> 2) + 2 * (r & 3) + 8 * g;   // even
    int N    = nT * 16 + ln;
    const float* s = (K < K0) ? src0 : src1;
    long kk = (K < K0) ? K : (K - K0);
    float a = s[kk * 2048 + N];
    float b = s[kk * 2048 + N + 2048];   // K+1, same source half (K0 even)
    dst[idx] = pack_bf16x2(a, b);
}

// ---------------------------------------------------------------------------
// One GEMM pass: acc[4][2] += (LDS h rows) @ (packed bf16 weights).
//  bbase = weights + lane*32 + w*2048 bytes (per-lane fragment base).
//  A fragment: two ds_load_b128 from hsh[ln][kc*32 + 8g (+16)].
//  B fragments: 16 non-temporal global_load_b128, immediate offsets.
// ---------------------------------------------------------------------------
template <int KCHUNKS>
__device__ __forceinline__ void gemm_pass(const char* __restrict__ bbase,
                                          const unsigned short (*hsh)[1024],
                                          int ln, int g, v8f acc[4][2]) {
#pragma unroll 2
    for (int kc = 0; kc < KCHUNKS; ++kc) {
        Frag a;
        a.q[0] = *(const nuint4*)&hsh[ln][kc * 32 + 8 * g];
        a.q[1] = *(const nuint4*)&hsh[ln][kc * 32 + 8 * g + 16];

        Frag bf[8];
#pragma unroll
        for (int gt = 0; gt < 4; ++gt)
#pragma unroll
            for (int ut = 0; ut < 2; ++ut) {
                const nuint4* p = (const nuint4*)(bbase + gt * 32768 + ut * 1024);
                bf[gt * 2 + ut].q[0] = __builtin_nontemporal_load(p);
                bf[gt * 2 + ut].q[1] = __builtin_nontemporal_load(p + 1);
            }
#pragma unroll
        for (int gt = 0; gt < 4; ++gt)
#pragma unroll
            for (int ut = 0; ut < 2; ++ut)
                acc[gt][ut] = __builtin_amdgcn_wmma_f32_16x16x32_bf16(
                    false, a.v, false, bf[gt * 2 + ut].v, (short)0,
                    acc[gt][ut], false, false);

        bbase += 128 * 1024;               // next k-chunk (128 tiles * 1 KB)
    }
}

// ---------------------------------------------------------------------------
// Persistent fused LSTM kernel. grid = 32 blocks, block = 512 threads (16 waves)
// ---------------------------------------------------------------------------
#define T_STEPS 168

__global__ __launch_bounds__(512, 1)
void pew_lstm_fused(const float* __restrict__ inputs,   // [512][168][5]
                    const float* __restrict__ W1,       // [2048]
                    const float* __restrict__ V1,       // [4][2048]
                    const float* __restrict__ b1,       // [2048]
                    const float* __restrict__ V2,       // [4][2048]
                    const float* __restrict__ b2,       // [2048]
                    const float* __restrict__ fc_w,     // [512]
                    const float* __restrict__ fc_b,     // [1]
                    const unsigned short* __restrict__ U1p,   // packed bf16 [512 x 2048]
                    const unsigned short* __restrict__ WU2p,  // packed bf16 [1024 x 2048]
                    float* __restrict__ out)            // [512*168]
{
    __shared__ __align__(16) unsigned short hsh[16][1024]; // [row][0:512)=h1, [512:1024)=h2
    __shared__ float wsh[16][5];                           // weather(4)+count per row

    const int tid     = threadIdx.x;
    const int lane    = tid & 31;
    const int w       = tid >> 5;          // wave 0..15 -> hidden units [w*32, w*32+32)
    const int g       = lane >> 4;         // half-wave
    const int ln      = lane & 15;
    const int rowBase = blockIdx.x * 16;   // batch rows owned by this block

    // per-lane fragment base pointers into the packed weights
    const char* bbase1 = (const char*)U1p  + lane * 32 + w * 2048;
    const char* bbase2 = (const char*)WU2p + lane * 32 + w * 2048;

    // zero h state (t=0: h_prev = 0)
    for (int i = tid; i < 16 * 1024 / 2; i += 512) ((unsigned*)hsh)[i] = 0u;
    __syncthreads();

    // c-state in registers: [unit-tile][C reg]; element = (row r+8g, unit ut*16+ln)
    float c1r[2][8], c2r[2][8];
#pragma unroll
    for (int ut = 0; ut < 2; ++ut)
#pragma unroll
        for (int r = 0; r < 8; ++r) { c1r[ut][r] = 0.0f; c2r[ut][r] = 0.0f; }

    // FC weights: lane holds fc_w[lane + 32k]
    float fcw[16];
#pragma unroll
    for (int k = 0; k < 16; ++k) fcw[k] = fc_w[lane + 32 * k];
    const float fcb = fc_b[0];

    const v8f zero8 = {0.f, 0.f, 0.f, 0.f, 0.f, 0.f, 0.f, 0.f};

#pragma unroll 1
    for (int t = 0; t < T_STEPS; ++t) {
        // stage weather + counts for this block's 16 rows
        if (tid < 80) {
            int rr = tid / 5, j = tid % 5;
            wsh[rr][j] = inputs[((long)(rowBase + rr) * T_STEPS + t) * 5 + j];
        }

        // ================= Layer 1: gates = h1_prev @ U1  (K = 512) =========
        v8f acc[4][2];
#pragma unroll
        for (int gt = 0; gt < 4; ++gt)
#pragma unroll
            for (int ut = 0; ut < 2; ++ut) acc[gt][ut] = zero8;

        gemm_pass<16>(bbase1, hsh, ln, g, acc);
        __syncthreads();   // all A reads done; wsh visible

        // LSTM cell, layer 1 (in registers)
        float h1v[2][8];
#pragma unroll
        for (int ut = 0; ut < 2; ++ut) {
            int nu = w * 32 + ut * 16 + ln;
            float w1g[4], b1g[4], v1g[4][4];
#pragma unroll
            for (int gt = 0; gt < 4; ++gt) {
                int n = gt * 512 + nu;
                w1g[gt] = W1[n]; b1g[gt] = b1[n];
#pragma unroll
                for (int j = 0; j < 4; ++j) v1g[gt][j] = V1[j * 2048 + n];
            }
#pragma unroll
            for (int r = 0; r < 8; ++r) {
                int row = r + 8 * g;
                float cnt = wsh[row][4];
                float gates[4];
#pragma unroll
                for (int gt = 0; gt < 4; ++gt) {
                    float p = acc[gt][ut][r] + cnt * w1g[gt] + b1g[gt];
#pragma unroll
                    for (int j = 0; j < 4; ++j) p += wsh[row][j] * v1g[gt][j];
                    gates[gt] = p;
                }
                float iv = sigmoid_f(gates[0]);
                float fv = sigmoid_f(gates[1]);
                float gv = tanh_f(gates[2]);
                float ov = sigmoid_f(gates[3]);
                float c  = fv * c1r[ut][r] + iv * gv;
                c1r[ut][r] = c;
                h1v[ut][r] = ov * tanh_f(c);
            }
        }
        // publish h1_t (LDS cols 0..511)
#pragma unroll
        for (int ut = 0; ut < 2; ++ut)
#pragma unroll
            for (int r = 0; r < 8; ++r)
                hsh[r + 8 * g][w * 32 + ut * 16 + ln] = f2bf(h1v[ut][r]);
        __syncthreads();

        // ========= Layer 2: gates = [h1_t | h2_prev] @ [W2;U2]  (K = 1024) ==
#pragma unroll
        for (int gt = 0; gt < 4; ++gt)
#pragma unroll
            for (int ut = 0; ut < 2; ++ut) acc[gt][ut] = zero8;

        gemm_pass<32>(bbase2, hsh, ln, g, acc);
        __syncthreads();   // all A reads (incl. h2_prev) done before overwrite

        // LSTM cell, layer 2
        float h2v[2][8];
#pragma unroll
        for (int ut = 0; ut < 2; ++ut) {
            int nu = w * 32 + ut * 16 + ln;
            float b2g[4], v2g[4][4];
#pragma unroll
            for (int gt = 0; gt < 4; ++gt) {
                int n = gt * 512 + nu;
                b2g[gt] = b2[n];
#pragma unroll
                for (int j = 0; j < 4; ++j) v2g[gt][j] = V2[j * 2048 + n];
            }
#pragma unroll
            for (int r = 0; r < 8; ++r) {
                int row = r + 8 * g;
                float gates[4];
#pragma unroll
                for (int gt = 0; gt < 4; ++gt) {
                    float p = acc[gt][ut][r] + b2g[gt];
#pragma unroll
                    for (int j = 0; j < 4; ++j) p += wsh[row][j] * v2g[gt][j];
                    gates[gt] = p;
                }
                float iv = sigmoid_f(gates[0]);
                float fv = sigmoid_f(gates[1]);
                float gv = tanh_f(gates[2]);
                float ov = sigmoid_f(gates[3]);
                float c  = fv * c2r[ut][r] + iv * gv;
                c2r[ut][r] = c;
                h2v[ut][r] = ov * tanh_f(c);
            }
        }
        // publish h2_t (LDS cols 512..1023)
#pragma unroll
        for (int ut = 0; ut < 2; ++ut)
#pragma unroll
            for (int r = 0; r < 8; ++r)
                hsh[r + 8 * g][512 + w * 32 + ut * 16 + ln] = f2bf(h2v[ut][r]);
        __syncthreads();

        // ================= FC head: out[b*T+t] = h2 . fc_w + fc_b ===========
        {
            int row = w;                      // wave w handles batch row rowBase+w
            float s = 0.0f;
#pragma unroll
            for (int k = 0; k < 16; ++k)
                s += bf2f(hsh[row][512 + lane + 32 * k]) * fcw[k];
#pragma unroll
            for (int off = 16; off >= 1; off >>= 1) s += __shfl_xor(s, off, 32);
            if (lane == 0)
                out[(long)(rowBase + row) * T_STEPS + t] = s + fcb;
        }
        // (next iteration's LDS writes are fenced by its own barriers)
    }
}

// ---------------------------------------------------------------------------
extern "C" void kernel_launch(void* const* d_in, const int* in_sizes, int n_in,
                              void* d_out, int out_size, void* d_ws, size_t ws_size,
                              hipStream_t stream) {
    (void)in_sizes; (void)n_in; (void)out_size; (void)ws_size;
    const float* inputs = (const float*)d_in[0];
    const float* W1     = (const float*)d_in[1];
    const float* U1     = (const float*)d_in[2];
    const float* V1     = (const float*)d_in[3];
    const float* b1     = (const float*)d_in[4];
    const float* W2     = (const float*)d_in[5];
    const float* U2     = (const float*)d_in[6];
    const float* V2     = (const float*)d_in[7];
    const float* b2     = (const float*)d_in[8];
    const float* fc_w   = (const float*)d_in[9];
    const float* fc_b   = (const float*)d_in[10];

    unsigned short* U1p  = (unsigned short*)d_ws;            // 512*2048 bf16 = 2 MB
    unsigned short* WU2p = U1p + (size_t)512 * 2048;         // 1024*2048 bf16 = 4 MB

    // Pack U1 -> bf16 fragment layout (K=512 -> 16 k-chunks)
    {
        long total = 16L * 128 * 32 * 8;                     // 524288 dwords
        int blocks = (int)((total + 255) / 256);
        pack_wmma_b<<<blocks, 256, 0, stream>>>(U1, 512, U1, (unsigned*)U1p, total);
    }
    // Pack [W2; U2] -> bf16 fragment layout (K=1024 -> 32 k-chunks)
    {
        long total = 32L * 128 * 32 * 8;                     // 1048576 dwords
        int blocks = (int)((total + 255) / 256);
        pack_wmma_b<<<blocks, 256, 0, stream>>>(W2, 512, U2, (unsigned*)WU2p, total);
    }

    pew_lstm_fused<<<32, 512, 0, stream>>>(inputs, W1, V1, b1, V2, b2,
                                           fc_w, fc_b, U1p, WU2p, (float*)d_out);
}
